// SMPL_54674933678685
// MI455X (gfx1250) — matrix operand order repaired and checked by reading, not compiled, not run on hardware
//
#include <hip/hip_runtime.h>
#include <hip/hip_bf16.h>
#include <stdint.h>
#include <stddef.h>

typedef __attribute__((ext_vector_type(16))) _Float16 v16h;
typedef __attribute__((ext_vector_type(8)))  _Float16 v8h;
typedef __attribute__((ext_vector_type(8)))  float    v8f;

#define B_    256
#define V_    6890
#define J_    24
#define NB    10
#define K_PF  207          // (J-1)*9
#define K_PAD 224          // 7 * 32 (WMMA K chunks)
#define N3    (V_*3)       // 20670
#define VT    16           // vertices per block tile (48 columns)

__constant__ int kParents[24] = {-1,0,0,0,1,2,3,4,5,6,7,8,9,9,9,12,13,14,16,17,18,19,20,21};

// ---------------------------------------------------------------------------
// K1: batch-independent contraction  Jbase = J_regressor @ v_template,
//     Jdirs = J_regressor @ shapedirs.  One block per (joint, coord) pair.
// ---------------------------------------------------------------------------
__global__ void __launch_bounds__(128)
jreg_reduce_kernel(const float* __restrict__ v_template,
                   const float* __restrict__ shapedirs,
                   const float* __restrict__ J_regressor,
                   float* __restrict__ Jbase,    // [72]
                   float* __restrict__ Jdirs)    // [720]
{
    const int jk  = blockIdx.x;           // 0..71
    const int j   = jk / 3, k = jk % 3;
    const int tid = threadIdx.x;          // 128
    float acc[11];
    #pragma unroll
    for (int i = 0; i < 11; ++i) acc[i] = 0.f;
    for (int v = tid; v < V_; v += 128) {
        const float w = J_regressor[j*V_ + v];
        acc[0] += w * v_template[v*3 + k];
        const float* sd = shapedirs + (size_t)(v*3 + k)*NB;
        #pragma unroll
        for (int l = 0; l < NB; ++l) acc[1+l] += w * sd[l];
    }
    __shared__ float red[11][128];
    #pragma unroll
    for (int i = 0; i < 11; ++i) red[i][tid] = acc[i];
    __syncthreads();
    for (int s = 64; s > 0; s >>= 1) {
        if (tid < s) {
            #pragma unroll
            for (int i = 0; i < 11; ++i) red[i][tid] += red[i][tid + s];
        }
        __syncthreads();
    }
    if (tid == 0) {
        Jbase[jk] = red[0][0];
        #pragma unroll
        for (int l = 0; l < NB; ++l) Jdirs[jk*NB + l] = red[1+l][0];
    }
}

// ---------------------------------------------------------------------------
// K2: posedirs [207][20670] f32  ->  Bt [20670][224] f16 (N-major, K zero-pad)
// ---------------------------------------------------------------------------
__global__ void __launch_bounds__(256)
posedirs_to_f16_kernel(const float* __restrict__ posedirs,
                       _Float16* __restrict__ Bt)
{
    const int n = blockIdx.x*256 + threadIdx.x;
    if (n >= N3) return;
    _Float16* dst = Bt + (size_t)n*K_PAD;
    for (int k = 0; k < K_PAD; ++k)
        dst[k] = (k < K_PF) ? (_Float16)posedirs[(size_t)k*N3 + n] : (_Float16)0.f;
}

// ---------------------------------------------------------------------------
// K3: per-batch wave32: Rodrigues, pose_feature (f16, padded), Jts,
//     kinematic chain, relative transforms A, posed joints -> d_out.
// ---------------------------------------------------------------------------
__global__ void __launch_bounds__(32)
pose_rigid_kernel(const float* __restrict__ betas,
                  const float* __restrict__ pose,
                  const float* __restrict__ Jbase,
                  const float* __restrict__ Jdirs,
                  _Float16* __restrict__ pfA,     // [256][224] f16
                  float* __restrict__ A_ws,       // [256][24][12]
                  float* __restrict__ Jt_out)     // [256][24][3] (in d_out)
{
    const int b    = blockIdx.x;
    const int lane = threadIdx.x;
    __shared__ float rotS[24][9];
    __shared__ float JtS[24][3];
    __shared__ float TfS[24][16];

    if (lane < 24) {
        float R[9] = {1.f,0.f,0.f, 0.f,1.f,0.f, 0.f,0.f,1.f};
        if (lane < 22) {   // pose[:, :-6] -> 22 axis-angle joints, 2 identity hands
            const float x = pose[b*72 + lane*3 + 0];
            const float y = pose[b*72 + lane*3 + 1];
            const float z = pose[b*72 + lane*3 + 2];
            const float xe = x + 1e-8f, ye = y + 1e-8f, ze = z + 1e-8f;
            const float ang = sqrtf(xe*xe + ye*ye + ze*ze);
            const float inv = 1.f / ang;
            const float rx = x*inv, ry = y*inv, rz = z*inv;
            const float s = sinf(ang), c = cosf(ang), o = 1.f - c;
            R[0] = c + o*rx*rx;      R[1] = o*rx*ry - s*rz;  R[2] = o*rx*rz + s*ry;
            R[3] = o*ry*rx + s*rz;   R[4] = c + o*ry*ry;     R[5] = o*ry*rz - s*rx;
            R[6] = o*rz*rx - s*ry;   R[7] = o*rz*ry + s*rx;  R[8] = c + o*rz*rz;
        }
        #pragma unroll
        for (int e = 0; e < 9; ++e) rotS[lane][e] = R[e];
        const float* be = betas + b*NB;
        #pragma unroll
        for (int k = 0; k < 3; ++k) {
            float acc = Jbase[lane*3 + k];
            const float* jd = Jdirs + (lane*3 + k)*NB;
            #pragma unroll
            for (int l = 0; l < NB; ++l) acc += be[l]*jd[l];
            JtS[lane][k] = acc;
        }
    }
    __syncthreads();

    // pose_feature = (rot[1:] - I) flattened, zero-padded to 224, as f16
    for (int idx = lane; idx < K_PAD; idx += 32) {
        float val = 0.f;
        if (idx < K_PF) {
            const int j = idx/9 + 1, e = idx % 9;
            val = rotS[j][e] - ((e % 4 == 0) ? 1.f : 0.f);
        }
        pfA[(size_t)b*K_PAD + idx] = (_Float16)val;
    }

    if (lane == 0) {   // sequential kinematic chain (depth <= 9)
        for (int i = 0; i < 24; ++i) {
            const int p = kParents[i];
            float M[16];
            #pragma unroll
            for (int r = 0; r < 3; ++r) {
                M[r*4+0] = rotS[i][r*3+0];
                M[r*4+1] = rotS[i][r*3+1];
                M[r*4+2] = rotS[i][r*3+2];
                M[r*4+3] = JtS[i][r] - (p >= 0 ? JtS[p][r] : 0.f);
            }
            M[12]=0.f; M[13]=0.f; M[14]=0.f; M[15]=1.f;
            if (p < 0) {
                #pragma unroll
                for (int t = 0; t < 16; ++t) TfS[i][t] = M[t];
            } else {
                for (int r = 0; r < 4; ++r)
                    for (int cc = 0; cc < 4; ++cc) {
                        float a = 0.f;
                        #pragma unroll
                        for (int t = 0; t < 4; ++t) a += TfS[p][r*4+t]*M[t*4+cc];
                        TfS[i][r*4+cc] = a;
                    }
            }
        }
    }
    __syncthreads();

    if (lane < 24) {
        const int i = lane;
        #pragma unroll
        for (int k = 0; k < 3; ++k)
            Jt_out[(b*24 + i)*3 + k] = TfS[i][k*4+3];
        #pragma unroll
        for (int m = 0; m < 3; ++m) {
            const float t = TfS[i][m*4+0]*JtS[i][0] + TfS[i][m*4+1]*JtS[i][1]
                          + TfS[i][m*4+2]*JtS[i][2];
            A_ws[(size_t)(b*24 + i)*12 + m*4+0] = TfS[i][m*4+0];
            A_ws[(size_t)(b*24 + i)*12 + m*4+1] = TfS[i][m*4+1];
            A_ws[(size_t)(b*24 + i)*12 + m*4+2] = TfS[i][m*4+2];
            A_ws[(size_t)(b*24 + i)*12 + m*4+3] = TfS[i][m*4+3] - t;
        }
    }
}

// ---------------------------------------------------------------------------
// K4 (fused): WMMA pose-offset GEMM + shaping epilogue -> LDS tile,
//             then LBS skinning in-block -> verts.  No v_posed round trip.
//
// Block = 3 waves (96 thr).  N footprint = 48 columns = 16 whole vertices;
// wave w owns the 16x16 C subtile at columns [v0*3 + 16w, +16).
// ---------------------------------------------------------------------------
__global__ void __launch_bounds__(96)
gemm_skin_kernel(const _Float16* __restrict__ pfA,       // [256][224]
                 const _Float16* __restrict__ Bt,        // [20670][224]
                 const float* __restrict__ betas,        // [256][10]
                 const float* __restrict__ v_template,   // [20670] flat
                 const float* __restrict__ shapedirs,    // [20670][10]
                 const float* __restrict__ lbs,          // [6890][24]
                 const float* __restrict__ A_ws,         // [256][24][12]
                 float* __restrict__ verts)              // [256][6890][3]
{
    const int tid  = threadIdx.x;          // 0..95
    const int wave = tid >> 5;             // 0..2
    const int lane = tid & 31;
    const int lm   = lane & 15;
    const bool hi  = lane >= 16;
    const int m0   = blockIdx.y * 16;      // batch tile
    const int v0   = blockIdx.x * VT;      // vertex tile
    const int nb   = v0 * 3;               // first column of block
    const int col  = nb + wave*16 + lm;    // this lane's column
    const int row  = m0 + lm;
    const bool colOK = (col < N3);

    __shared__ float posedS[16][VT*3];     // [batch-row][coord-in-tile]  3 KB
    __shared__ float AS[16][24*12];        // rel transforms for 16 batches 18 KB
    __shared__ float WS[VT][24];           // lbs rows for 16 vertices   1.5 KB

    // ---- WMMA GEMM over K=224 ------------------------------------------
    const _Float16* Arow = pfA + (size_t)row * K_PAD;
    const _Float16* Brow = Bt  + (size_t)(colOK ? col : 0) * K_PAD;  // branch-free;
                                    // OOB lane only pollutes its own unstored C column
    v8f c = {};
    #pragma unroll
    for (int kt = 0; kt < K_PAD/32; ++kt) {
        const int k0 = kt*32;
        if (kt + 1 < K_PAD/32)
            __builtin_prefetch((const void*)(Brow + k0 + 32), 0, 1);  // global_prefetch_b8
        union { v16h v; v8h h[2]; } a, bf;
        // 16-bit A layout: lanes 0-15 hold K {0..7,16..23}, lanes 16-31 {8..15,24..31}
        const int ab = k0 + (hi ? 8 : 0);
        a.h[0] = *(const v8h*)(Arow + ab);
        a.h[1] = *(const v8h*)(Arow + ab + 16);
        // 16-bit B layout: same K order as A: lanes 0-15 hold K {0..7,16..23}, lanes 16-31 {8..15,24..31}
        const int bb = k0 + (hi ? 8 : 0);
        bf.h[0] = *(const v8h*)(Brow + bb);
        bf.h[1] = *(const v8h*)(Brow + bb + 16);
        c = __builtin_amdgcn_wmma_f32_16x16x32_f16(false, a.v, false, bf.v,
                                                   (short)0, c, false, false);
    }

    // ---- epilogue: + v_template + shapedirs.betas, deposit tile in LDS --
    if (colOK) {
        const float tmpl = v_template[col];
        float sd[NB];
        const float* sdp = shapedirs + (size_t)col*NB;
        #pragma unroll
        for (int l = 0; l < NB; ++l) sd[l] = sdp[l];
        const int lc = col - nb;           // 0..47 within block
        #pragma unroll
        for (int r = 0; r < 8; ++r) {
            const int rloc = r + (hi ? 8 : 0);   // C layout: VGPR r -> M = r (+8 hi half)
            const float* be = betas + (m0 + rloc)*NB;
            float val = tmpl;
            #pragma unroll
            for (int l = 0; l < NB; ++l) val += be[l]*sd[l];
            posedS[rloc][lc] = c[r] + val;
        }
    }

    // ---- stage A (16 batches) and lbs weight rows (16 verts) ------------
    for (int i = tid; i < 16*288; i += 96)
        AS[i/288][i%288] = A_ws[(size_t)(m0 + i/288)*288 + (i%288)];
    for (int i = tid; i < VT*24; i += 96) {
        const int vv = v0 + i/24;
        WS[i/24][i%24] = (vv < V_) ? lbs[(size_t)vv*24 + (i%24)] : 0.f;
    }
    __syncthreads();

    // ---- skinning: 16 batches x 16 verts = 256 pairs over 96 threads ----
    for (int idx = tid; idx < 16*VT; idx += 96) {
        const int br = idx >> 4;           // batch row in tile
        const int vr = idx & 15;           // vertex in tile
        const int v  = v0 + vr;
        if (v >= V_) continue;
        const float p0 = posedS[br][vr*3+0];
        const float p1 = posedS[br][vr*3+1];
        const float p2 = posedS[br][vr*3+2];
        float a0 = 0.f, a1 = 0.f, a2 = 0.f;
        #pragma unroll
        for (int j = 0; j < 24; ++j) {
            const float w = WS[vr][j];
            const float* Aj = &AS[br][j*12];
            a0 += w*(Aj[0]*p0 + Aj[1]*p1 + Aj[2] *p2 + Aj[3]);
            a1 += w*(Aj[4]*p0 + Aj[5]*p1 + Aj[6] *p2 + Aj[7]);
            a2 += w*(Aj[8]*p0 + Aj[9]*p1 + Aj[10]*p2 + Aj[11]);
        }
        float* o = verts + ((size_t)(m0 + br)*V_ + v)*3;
        o[0] = a0; o[1] = a1; o[2] = a2;
    }
}

// ---------------------------------------------------------------------------
extern "C" void kernel_launch(void* const* d_in, const int* in_sizes, int n_in,
                              void* d_out, int out_size, void* d_ws, size_t ws_size,
                              hipStream_t stream) {
    const float* betas      = (const float*)d_in[0];   // [256,10]
    const float* pose       = (const float*)d_in[1];   // [256,72]
    const float* v_template = (const float*)d_in[2];   // [6890,3]
    const float* shapedirs  = (const float*)d_in[3];   // [6890,3,10]
    const float* posedirs   = (const float*)d_in[4];   // [207, 20670]
    const float* J_reg      = (const float*)d_in[5];   // [24, 6890]
    const float* lbs        = (const float*)d_in[6];   // [6890, 24]
    (void)in_sizes; (void)n_in; (void)out_size;        // parents hardcoded (module const)

    // workspace layout (256-byte aligned chunks), ~9.7 MB total
    char* ws = (char*)d_ws;
    size_t off = 0;
    _Float16* Bt    = (_Float16*)(ws + off); off += ((size_t)N3*K_PAD*2 + 255) & ~(size_t)255;  // 9.26 MB
    _Float16* pfA   = (_Float16*)(ws + off); off += ((size_t)B_*K_PAD*2 + 255) & ~(size_t)255;  // 112 KB
    float*    Jbase = (float*)(ws + off);    off += 512;                                         // 72 f32
    float*    Jdirs = (float*)(ws + off);    off += 3072;                                        // 720 f32
    float*    A_ws  = (float*)(ws + off);    off += ((size_t)B_*288*4 + 255) & ~(size_t)255;     // 288 KB
    (void)ws_size; (void)off;

    float* verts  = (float*)d_out;                        // [256,6890,3]
    float* Jt_out = (float*)d_out + (size_t)B_*V_*3;      // [256,24,3]

    jreg_reduce_kernel<<<72, 128, 0, stream>>>(v_template, shapedirs, J_reg, Jbase, Jdirs);
    posedirs_to_f16_kernel<<<(N3 + 255)/256, 256, 0, stream>>>(posedirs, Bt);
    pose_rigid_kernel<<<B_, 32, 0, stream>>>(betas, pose, Jbase, Jdirs, pfA, A_ws, Jt_out);
    gemm_skin_kernel<<<dim3((V_ + VT - 1)/VT, B_/16), 96, 0, stream>>>(
        pfA, Bt, betas, v_template, shapedirs, lbs, A_ws, verts);
}
